// GlobalLocalCrossAttention_39702677684504
// MI455X (gfx1250) — matrix operand order repaired and checked by reading, not compiled, not run on hardware
//
#include <hip/hip_runtime.h>

#define NTOK   577
#define KEEP   58
#define BATCH  4
#define HEADS  12
#define LAYERS 12
#define CDIM   768
#define HDIM   64
#define NSEL   57

#define MX   (BATCH * NTOK)      // 2308 rows for x-based GEMMs
#define MXP  2320                // padded to %16
#define MS   (BATCH * NSEL)      // 228 rows for sel-based GEMMs
#define MSP  240                 // padded to %16

typedef __attribute__((ext_vector_type(16))) _Float16 v16h;
typedef __attribute__((ext_vector_type(8)))  _Float16 v8h;
typedef __attribute__((ext_vector_type(8)))  float    v8f;

// ---------------------------------------------------------------------------
// init: rollout vector v = e0 per batch, inverse map = -1, union flags = 0
// ---------------------------------------------------------------------------
__global__ void init_state(float* __restrict__ v, int* __restrict__ map,
                           int* __restrict__ uni) {
  int i = blockIdx.x * blockDim.x + threadIdx.x;
  if (i < BATCH * NTOK) {
    v[i]   = ((i % NTOK) == 0) ? 1.0f : 0.0f;
    map[i] = -1;
  }
  if (i < NTOK) uni[i] = 0;
}

// ---------------------------------------------------------------------------
// f32 -> f16 with row padding (pad rows zeroed; no OOB reads of src)
// ---------------------------------------------------------------------------
__global__ void cvt_f16_pad(const float* __restrict__ src, _Float16* __restrict__ dst,
                            int rows, int rowsPad, int cols) {
  size_t i = (size_t)blockIdx.x * blockDim.x + threadIdx.x;
  size_t total = (size_t)rowsPad * cols;
  if (i >= total) return;
  int r = (int)(i / cols);
  dst[i] = (r < rows) ? (_Float16)src[i] : (_Float16)0.0f;
}

// ---------------------------------------------------------------------------
// build a_l[b,j,:] : head-mean, top-KEEP mask (binary-search threshold),
// keep col 0, a = (masked + I)*0.5, row-normalize. One block per (row j, b).
// This is the single mandatory streaming pass over attention_history.
// ---------------------------------------------------------------------------
__global__ void build_a_kernel(const float* __restrict__ attn,
                               float* __restrict__ aBuf, int layer) {
  int j = blockIdx.x;
  int b = blockIdx.y;
  __shared__ float fused[NTOK];
  __shared__ float red[256];
  __shared__ int   cnt;
  const size_t NN = (size_t)NTOK * NTOK;
  const float* base =
      attn + ((((size_t)layer * BATCH + b) * HEADS) * NTOK + (size_t)j) * NTOK;

  for (int k = threadIdx.x; k < NTOK; k += 256) {
    float s = 0.0f;
#pragma unroll
    for (int h = 0; h < HEADS; ++h) s += base[(size_t)h * NN + k];
    fused[k] = s * (1.0f / (float)HEADS);
  }
  __syncthreads();

  // row max (values are softmax means, >= 0)
  float lm = 0.0f;
  for (int k = threadIdx.x; k < NTOK; k += 256) lm = fmaxf(lm, fused[k]);
  red[threadIdx.x] = lm;
  __syncthreads();
  for (int off = 128; off > 0; off >>= 1) {
    if (threadIdx.x < off)
      red[threadIdx.x] = fmaxf(red[threadIdx.x], red[threadIdx.x + off]);
    __syncthreads();
  }
  float hi = red[0] * 1.000001f + 1e-12f;
  float lo = 0.0f;
  __syncthreads();

  // binary search for the KEEP-th largest value
  for (int it = 0; it < 24; ++it) {
    float mid = 0.5f * (lo + hi);
    if (threadIdx.x == 0) cnt = 0;
    __syncthreads();
    int c = 0;
    for (int k = threadIdx.x; k < NTOK; k += 256) c += (fused[k] >= mid) ? 1 : 0;
    if (c) atomicAdd(&cnt, c);
    __syncthreads();
    if (cnt >= KEEP) lo = mid; else hi = mid;
    __syncthreads();
  }
  float tau = lo;

  // row sum of a = (masked + eye_j) * 0.5
  float ls = 0.0f;
  for (int k = threadIdx.x; k < NTOK; k += 256) {
    float f = fused[k];
    float m = (f >= tau || k == 0) ? f : 0.0f;
    ls += m;
  }
  red[threadIdx.x] = ls;
  __syncthreads();
  for (int off = 128; off > 0; off >>= 1) {
    if (threadIdx.x < off) red[threadIdx.x] += red[threadIdx.x + off];
    __syncthreads();
  }
  float inv = 1.0f / (0.5f * (red[0] + 1.0f));  // +1 from the diagonal

  float* arow = aBuf + ((size_t)b * NTOK + j) * NTOK;
  for (int k = threadIdx.x; k < NTOK; k += 256) {
    float f = fused[k];
    float m = (f >= tau || k == 0) ? f : 0.0f;
    float a = (m + ((k == j) ? 1.0f : 0.0f)) * 0.5f;
    arow[k] = a * inv;
  }
}

// ---------------------------------------------------------------------------
// rollout GEMV: vout[b,k] = sum_j vin[b,j] * a[b,j,k]   (coalesced over k)
// ---------------------------------------------------------------------------
__global__ void rollout_gemv(const float* __restrict__ aBuf,
                             const float* __restrict__ vin,
                             float* __restrict__ vout) {
  int b = blockIdx.y;
  int k = blockIdx.x * blockDim.x + threadIdx.x;
  __shared__ float vs[NTOK];
  for (int j = threadIdx.x; j < NTOK; j += blockDim.x)
    vs[j] = vin[b * NTOK + j];
  __syncthreads();
  if (k >= NTOK) return;
  const float* a = aBuf + (size_t)b * NTOK * NTOK + k;
  float acc = 0.0f;
  for (int j = 0; j < NTOK; ++j) {
    if (j + 8 < NTOK) __builtin_prefetch(a + (size_t)(j + 8) * NTOK, 0, 0);
    acc += vs[j] * a[(size_t)j * NTOK];
  }
  vout[b * NTOK + k] = acc;
}

// ---------------------------------------------------------------------------
// ordered top-NSEL per batch (exact, score-descending like jax top_k);
// also fills per-batch inverse map and the cross-batch union flags.
// ---------------------------------------------------------------------------
__global__ void topk_sel(const float* __restrict__ v, int* __restrict__ sel,
                         int* __restrict__ map, int* __restrict__ uni) {
  int b = blockIdx.x;
  __shared__ float sc[NTOK - 1];
  __shared__ float rmax[256];
  __shared__ int   ridx[256];
  for (int i = threadIdx.x; i < NTOK - 1; i += blockDim.x)
    sc[i] = v[b * NTOK + 1 + i];
  __syncthreads();
  for (int s = 0; s < NSEL; ++s) {
    float bm = -1e30f; int bi = -1;
    for (int i = threadIdx.x; i < NTOK - 1; i += blockDim.x)
      if (sc[i] > bm) { bm = sc[i]; bi = i; }
    rmax[threadIdx.x] = bm; ridx[threadIdx.x] = bi;
    __syncthreads();
    for (int off = 128; off > 0; off >>= 1) {
      if (threadIdx.x < off) {
        if (rmax[threadIdx.x + off] > rmax[threadIdx.x]) {
          rmax[threadIdx.x] = rmax[threadIdx.x + off];
          ridx[threadIdx.x] = ridx[threadIdx.x + off];
        }
      }
      __syncthreads();
    }
    if (threadIdx.x == 0) {
      int tok = ridx[0] + 1;               // shift past CLS
      sel[b * NSEL + s]   = tok;
      map[b * NTOK + tok] = s;
      uni[tok] = 1;                        // benign race (all write 1)
      sc[ridx[0]] = -1e30f;
    }
    __syncthreads();
  }
}

// ---------------------------------------------------------------------------
// gather selected query tokens directly into padded f16 GEMM input
// ---------------------------------------------------------------------------
__global__ void gather_lq(const float* __restrict__ x, const int* __restrict__ sel,
                          _Float16* __restrict__ lq) {
  int bs = blockIdx.x;
  int b = bs / NSEL, s = bs % NSEL;
  int tok = sel[b * NSEL + s];
  const float* src = x + ((size_t)b * NTOK + tok) * CDIM;
  _Float16* dst = lq + (size_t)bs * CDIM;
  for (int c = threadIdx.x; c < CDIM; c += blockDim.x)
    dst[c] = (_Float16)src[c];
}

// ---------------------------------------------------------------------------
// WMMA GEMM (f16 inputs, f32 accumulate): C[Mp,N] = A[Mp,K] @ W^T + bias
// A: [Mp,K] f16 row-major (Mp %16==0), W: [N,K] f16 row-major, C padded.
// One wave = one 16x16 tile. Inner loop: 4x global_load_b128 + 1x v_wmma.
// A frag: lanes 0-15 hold K 0..15, lanes 16-31 hold K 16..31.
// B frag: lane = column; lanes 0-15 hold K 0..15, lanes 16-31 hold K 16..31.
// ---------------------------------------------------------------------------
__global__ void gemm_f16_wmma(const _Float16* __restrict__ A,
                              const _Float16* __restrict__ W,
                              const float* __restrict__ bias,
                              float* __restrict__ Cm, int Mp, int N, int K) {
  int wave = blockIdx.x * (blockDim.x >> 5) + (threadIdx.x >> 5);
  int lane = threadIdx.x & 31;
  int tilesN = N >> 4;
  int tm = wave / tilesN;
  int tn = wave - tm * tilesN;
  if (tm * 16 >= Mp) return;                      // wave-uniform exit

  int row = tm * 16 + (lane & 15);
  int col = tn * 16 + (lane & 15);
  int ka = (lane < 16) ? 0 : 16;
  int kb = (lane < 16) ? 0 : 16;
  const _Float16* ap0 = A + (size_t)row * K + ka;
  const _Float16* wp0 = W + (size_t)col * K + kb;

  v8f acc = {};
#pragma unroll 4
  for (int k0 = 0; k0 < K; k0 += 32) {
    v8h  alo = *(const v8h*)(ap0 + k0);
    v8h  ahi = *(const v8h*)(ap0 + k0 + 8);
    v16h af  = __builtin_shufflevector(alo, ahi, 0, 1, 2, 3, 4, 5, 6, 7,
                                       8, 9, 10, 11, 12, 13, 14, 15);
    v16h bf  = *(const v16h*)(wp0 + k0);
    acc = __builtin_amdgcn_wmma_f32_16x16x32_f16(false, af, false, bf,
                                                 (short)0, acc, false, false);
  }

  int mBase = tm * 16 + ((lane >= 16) ? 8 : 0);
  float bv = bias[col];
  float* cp = Cm + (size_t)mBase * N + col;
#pragma unroll
  for (int i = 0; i < 8; ++i)
    cp[(size_t)i * N] = acc[i] + bv;              // Mp padded: no guards
}

// ---------------------------------------------------------------------------
// attention core: one block per (s, h, b). logits via wave-per-key-row dot
// (shuffle reduce), block softmax, attn_w written in top-k order, then w @ V.
// K/V total ~14 MB -> resident in the 192 MB L2 across all 2736 blocks.
// ---------------------------------------------------------------------------
__global__ void attn_core(const float* __restrict__ qbuf, const float* __restrict__ kbuf,
                          const float* __restrict__ vbuf, float* __restrict__ attn_w,
                          float* __restrict__ attnOut) {
  int s = blockIdx.x, h = blockIdx.y, b = blockIdx.z;
  __shared__ float qs[HDIM];
  __shared__ float w[NTOK];
  __shared__ float red[256];
  __shared__ float oacc[4][HDIM];
  int tid = threadIdx.x;

  const float* qp = qbuf + ((size_t)(b * NSEL + s)) * CDIM + h * HDIM;
  if (tid < HDIM) qs[tid] = qp[tid];
  __syncthreads();

  int wid = tid >> 5, lane = tid & 31;
  for (int n = wid; n < NTOK; n += 8) {
    const float* kp = kbuf + ((size_t)b * NTOK + n) * CDIM + h * HDIM;
    float p = qs[lane] * kp[lane] + qs[lane + 32] * kp[lane + 32];
#pragma unroll
    for (int off = 16; off > 0; off >>= 1) p += __shfl_xor(p, off, 32);
    if (lane == 0) w[n] = p * 0.125f;   // scale = 64^-0.5
  }
  __syncthreads();

  // softmax max
  float lm = -3.0e38f;
  for (int n = tid; n < NTOK; n += 256) lm = fmaxf(lm, w[n]);
  red[tid] = lm;
  __syncthreads();
  for (int off = 128; off > 0; off >>= 1) {
    if (tid < off) red[tid] = fmaxf(red[tid], red[tid + off]);
    __syncthreads();
  }
  float mx = red[0];
  __syncthreads();

  // exp + sum
  float lsum = 0.0f;
  for (int n = tid; n < NTOK; n += 256) {
    float e = __expf(w[n] - mx);
    w[n] = e;
    lsum += e;
  }
  red[tid] = lsum;
  __syncthreads();
  for (int off = 128; off > 0; off >>= 1) {
    if (tid < off) red[tid] += red[tid + off];
    __syncthreads();
  }
  float inv = 1.0f / red[0];

  float* awp = attn_w + (((size_t)(b * HEADS + h) * NSEL + s)) * NTOK;
  for (int n = tid; n < NTOK; n += 256) {
    float wn = w[n] * inv;
    w[n] = wn;
    awp[n] = wn;
  }
  __syncthreads();

  // out = w @ V  (threads: d = tid&63, 4 partial sums over n)
  int d = tid & 63, part = tid >> 6;
  float acc = 0.0f;
  for (int n = part; n < NTOK; n += 4)
    acc += w[n] * vbuf[((size_t)b * NTOK + n) * CDIM + h * HDIM + d];
  oacc[part][d] = acc;
  __syncthreads();
  if (tid < HDIM) {
    float o = oacc[0][tid] + oacc[1][tid] + oacc[2][tid] + oacc[3][tid];
    attnOut[((size_t)(b * NSEL + s)) * CDIM + h * HDIM + tid] = o;
  }
}

// ---------------------------------------------------------------------------
// final scatter: CLS -> x, selected -> proj row, union-but-not-selected -> 0,
// never-selected -> x
// ---------------------------------------------------------------------------
__global__ void scatter_out(const float* __restrict__ x, const float* __restrict__ proj,
                            const int* __restrict__ map, const int* __restrict__ uni,
                            float* __restrict__ out) {
  int bn = blockIdx.x;
  int b = bn / NTOK, n = bn % NTOK;
  float* dst = out + (size_t)bn * CDIM;
  const float* xs = x + (size_t)bn * CDIM;
  int s = map[b * NTOK + n];
  if (n == 0) {
    for (int c = threadIdx.x; c < CDIM; c += blockDim.x) dst[c] = xs[c];
  } else if (s >= 0) {
    const float* pr = proj + ((size_t)(b * NSEL + s)) * CDIM;
    for (int c = threadIdx.x; c < CDIM; c += blockDim.x) dst[c] = pr[c];
  } else if (!uni[n]) {
    for (int c = threadIdx.x; c < CDIM; c += blockDim.x) dst[c] = xs[c];
  } else {
    for (int c = threadIdx.x; c < CDIM; c += blockDim.x) dst[c] = 0.0f;
  }
}

// ---------------------------------------------------------------------------
static inline void launch_gemm(const _Float16* A, const _Float16* W, const float* bias,
                               float* C, int Mp, int N, int K, hipStream_t stream) {
  int tiles = (Mp / 16) * (N / 16);
  int blocks = (tiles + 7) / 8;  // 8 waves (256 threads) per block
  gemm_f16_wmma<<<blocks, 256, 0, stream>>>(A, W, bias, C, Mp, N, K);
}

static inline void launch_cvt(const float* src, _Float16* dst, int rows, int rowsPad,
                              int cols, hipStream_t stream) {
  size_t total = (size_t)rowsPad * cols;
  cvt_f16_pad<<<(int)((total + 255) / 256), 256, 0, stream>>>(src, dst, rows, rowsPad, cols);
}

extern "C" void kernel_launch(void* const* d_in, const int* in_sizes, int n_in,
                              void* d_out, int out_size, void* d_ws, size_t ws_size,
                              hipStream_t stream) {
  const float* x  = (const float*)d_in[0];
  const float* ah = (const float*)d_in[1];
  const float* Wq = (const float*)d_in[2]; const float* bq = (const float*)d_in[3];
  const float* Wk = (const float*)d_in[4]; const float* bk = (const float*)d_in[5];
  const float* Wv = (const float*)d_in[6]; const float* bv = (const float*)d_in[7];
  const float* Wo = (const float*)d_in[8]; const float* bo = (const float*)d_in[9];

  float* out   = (float*)d_out;                                  // [B,N,C]
  float* attnW = out + (size_t)BATCH * NTOK * CDIM;              // [B,H,S,N]

  // workspace carve-out (~31 MB total)
  char* ws = (char*)d_ws;
  auto alloc = [&](size_t bytes) -> char* {
    char* p = ws;
    ws += (bytes + 255) & ~(size_t)255;
    return p;
  };
  float*    aBuf  = (float*)   alloc((size_t)BATCH * NTOK * NTOK * 4);
  float*    vA    = (float*)   alloc((size_t)BATCH * NTOK * 4);
  float*    vB    = (float*)   alloc((size_t)BATCH * NTOK * 4);
  int*      sel   = (int*)     alloc((size_t)BATCH * NSEL * 4);
  int*      map   = (int*)     alloc((size_t)BATCH * NTOK * 4);
  int*      uni   = (int*)     alloc((size_t)NTOK * 4);
  _Float16* xH    = (_Float16*)alloc((size_t)MXP * CDIM * 2);
  _Float16* WqH   = (_Float16*)alloc((size_t)CDIM * CDIM * 2);
  _Float16* WkH   = (_Float16*)alloc((size_t)CDIM * CDIM * 2);
  _Float16* WvH   = (_Float16*)alloc((size_t)CDIM * CDIM * 2);
  _Float16* WoH   = (_Float16*)alloc((size_t)CDIM * CDIM * 2);
  _Float16* lqH   = (_Float16*)alloc((size_t)MSP * CDIM * 2);
  _Float16* aOutH = (_Float16*)alloc((size_t)MSP * CDIM * 2);
  float*    kB    = (float*)   alloc((size_t)MXP * CDIM * 4);
  float*    vBf   = (float*)   alloc((size_t)MXP * CDIM * 4);
  float*    qB    = (float*)   alloc((size_t)MSP * CDIM * 4);
  float*    aOut  = (float*)   alloc((size_t)MS  * CDIM * 4);
  float*    pOut  = (float*)   alloc((size_t)MSP * CDIM * 4);

  init_state<<<(BATCH * NTOK + 255) / 256, 256, 0, stream>>>(vA, map, uni);

  // one-time f32 -> f16 conversions (weights + x), padded rows zeroed
  launch_cvt(x,  xH,  MX,   MXP,  CDIM, stream);
  launch_cvt(Wq, WqH, CDIM, CDIM, CDIM, stream);
  launch_cvt(Wk, WkH, CDIM, CDIM, CDIM, stream);
  launch_cvt(Wv, WvH, CDIM, CDIM, CDIM, stream);
  launch_cvt(Wo, WoH, CDIM, CDIM, CDIM, stream);

  // rollout: row 0 of a_{L-1}..a_0 chain via GEMV chain (reverse layer order)
  float* vin = vA;
  float* vout = vB;
  for (int l = LAYERS - 1; l >= 0; --l) {
    build_a_kernel<<<dim3(NTOK, BATCH), 256, 0, stream>>>(ah, aBuf, l);
    rollout_gemv<<<dim3((NTOK + 255) / 256, BATCH), 256, 0, stream>>>(aBuf, vin, vout);
    float* t = vin; vin = vout; vout = t;
  }
  // vin now holds the CLS rollout row per batch; scores = vin[b][1..576]

  topk_sel<<<BATCH, 256, 0, stream>>>(vin, sel, map, uni);
  gather_lq<<<BATCH * NSEL, 256, 0, stream>>>(x, sel, lqH);

  // WMMA projections (f16 in, f32 out)
  launch_gemm(xH,  WkH, bk, kB,  MXP, CDIM, CDIM, stream);
  launch_gemm(xH,  WvH, bv, vBf, MXP, CDIM, CDIM, stream);
  launch_gemm(lqH, WqH, bq, qB,  MSP, CDIM, CDIM, stream);

  attn_core<<<dim3(NSEL, HEADS, BATCH), 256, 0, stream>>>(qB, kB, vBf, attnW, aOut);

  launch_cvt(aOut, aOutH, MS, MSP, CDIM, stream);
  launch_gemm(aOutH, WoH, bo, pOut, MSP, CDIM, CDIM, stream);

  scatter_out<<<BATCH * NTOK, 256, 0, stream>>>(x, pOut, map, uni, out);
}
